// LSTMSequenceModel_65859028517120
// MI455X (gfx1250) — matrix operand order repaired and checked by reading, not compiled, not run on hardware
//
#include <hip/hip_runtime.h>

typedef __attribute__((ext_vector_type(16))) _Float16 v16h;
typedef __attribute__((ext_vector_type(8)))  float    v8f;

#define Bn 2048
#define Tn 512
#define Dn 4
#define Hn 16
#define LDS_STRIDE 40   // halfs per row: 80 bytes -> 16B aligned chunks, no bank conflicts

__device__ __forceinline__ float fast_rcp(float x) {
    return __builtin_amdgcn_rcpf(x);   // v_rcp_f32, no div fixup chain
}

#if __has_builtin(__builtin_amdgcn_tanhf)
__device__ __forceinline__ float ftanh_fast(float x) {
    return __builtin_amdgcn_tanhf(x);          // native v_tanh_f32
}
#else
__device__ __forceinline__ float ftanh_fast(float x) {
    // tanh(x) = 1 - 2/(e^{2x} + 1)
    return __builtin_fmaf(-2.0f, fast_rcp(1.0f + __expf(2.0f * x)), 1.0f);
}
#endif

// Sigmoid of gate value g, given y = g/2 (the /2 is pre-folded into the
// WMMA weights/biases for the i, f, o gate tiles):
// sigmoid(g) = 0.5 + 0.5*tanh(g/2) = fma(0.5, tanh(y), 0.5)
__device__ __forceinline__ float fsigmoid_half(float y) {
    return __builtin_fmaf(0.5f, ftanh_fast(y), 0.5f);
}

__global__ __launch_bounds__(32, 1)
void lstm2_fused_kernel(const float* __restrict__ x,
                        const float* __restrict__ Wih0, const float* __restrict__ Whh0,
                        const float* __restrict__ bih0, const float* __restrict__ bhh0,
                        const float* __restrict__ Wih1, const float* __restrict__ Whh1,
                        const float* __restrict__ bih1, const float* __restrict__ bhh1,
                        const float* __restrict__ Wfc,  const float* __restrict__ bfc,
                        float* __restrict__ out)
{
    // A-operand staging buffers (f16), one 16x32 logical tile each (padded rows).
    // g0A columns: [0:4) = x_t, [4:20) = h0[t-1], [20:32) = zero pad
    // g1A columns: [0:16) = h0[t], [16:32) = h1[t-1]
    __shared__ _Float16 g0A[16][LDS_STRIDE];
    __shared__ _Float16 g1A[16][LDS_STRIDE];

    const int lane = threadIdx.x;
    const int half = lane >> 4;   // which 16-lane half of the wave
    const int lq   = lane & 15;
    const int b0   = blockIdx.x * 16;

    // Zero LDS (h0[-1] = h1[-1] = 0, and permanent K-padding zeros).
    for (int i = lane; i < 16 * LDS_STRIDE; i += 32) {
        (&g0A[0][0])[i] = (_Float16)0.0f;
        (&g1A[0][0])[i] = (_Float16)0.0f;
    }

    // ---- Preload weight B-fragments (resident in VGPRs for the whole run) ----
    // B (32x16 f16) lane layout: value j<8 of lane = B[half*8 + j][tile*16 + lq], j>=8 -> B[16 + half*8 + (j-8)][tile*16 + lq]
    // Layer0 concatenated B rows: k<4 -> Wih0[:,k]; 4<=k<20 -> Whh0[:,k-4]; else 0
    // Layer1 concatenated B rows: k<16 -> Wih1[:,k]; k>=16 -> Whh1[:,k-16]
    // Gate tiles 0,1,3 (i,f,o feed sigmoid) are pre-scaled by 0.5 so the WMMA
    // emits gate/2 directly into v_tanh_f32 (no per-step pre-scale VALU ops).
    v16h Wb0[4], Wb1[4];
    #pragma unroll
    for (int tile = 0; tile < 4; ++tile) {
        const float gscale = (tile == 2) ? 1.0f : 0.5f;
        const int col = tile * 16 + lq;
        #pragma unroll
        for (int j = 0; j < 16; ++j) {
            const int k = (j < 8) ? (half * 8 + j) : (16 + half * 8 + (j - 8));
            float w0 = (k < 4)  ? Wih0[col * Dn + k]
                     : (k < 20) ? Whh0[col * Hn + (k - 4)]
                                : 0.0f;
            float w1 = (k < 16) ? Wih1[col * Hn + k]
                                : Whh1[col * Hn + (k - 16)];
            Wb0[tile][j] = (_Float16)(w0 * gscale);
            Wb1[tile][j] = (_Float16)(w1 * gscale);
        }
    }

    // Per-column gate biases (bih + bhh), broadcast once into loop-invariant
    // accumulator-init vectors (WMMA C operand stays in fixed registers).
    v8f bias0v[4], bias1v[4];
    #pragma unroll
    for (int tile = 0; tile < 4; ++tile) {
        const float gscale = (tile == 2) ? 1.0f : 0.5f;
        const float bv0 = (bih0[tile * 16 + lq] + bhh0[tile * 16 + lq]) * gscale;
        const float bv1 = (bih1[tile * 16 + lq] + bhh1[tile * 16 + lq]) * gscale;
        #pragma unroll
        for (int v = 0; v < 8; ++v) { bias0v[tile][v] = bv0; bias1v[tile][v] = bv1; }
    }

    // FC weights (wave-uniform -> scalar regs).
    float wfcv[16];
    #pragma unroll
    for (int k = 0; k < 16; ++k) wfcv[k] = Wfc[k];
    const float bfcv = bfc[0];

    // Cell states in C/D register layout: value v -> row M = v + half*8, col N = lq.
    float c0s[8], c1s[8];
    #pragma unroll
    for (int v = 0; v < 8; ++v) { c0s[v] = 0.0f; c1s[v] = 0.0f; }

    // Per-lane x source pointer: lane covers row lq, columns half*2 .. half*2+1.
    const float* xbase = x + ((size_t)(b0 + lq) * Tn) * Dn + half * 2;

    // Prime the x pipeline (prefetch depth 1: HBM latency hides under a full
    // timestep of recurrence work instead of sitting on the critical path).
    float2 xv = *(const float2*)(xbase);

    for (int t = 0; t < Tn; ++t) {
        // ---- stage x_t (prefetched last iteration) into g0A cols 0..3 ----
        g0A[lq][half * 2 + 0] = (_Float16)xv.x;
        g0A[lq][half * 2 + 1] = (_Float16)xv.y;

        // ---- issue prefetch of x_{t+1} (clamped; waited next iteration) ----
        const int tn = (t + 1 < Tn) ? (t + 1) : t;
        const float2 xv_next = *(const float2*)(xbase + (size_t)tn * Dn);

        // ---- layer 0: gates0 = [x_t | h0_prev] @ [Wih0; Whh0]^T + bias ----
        // A (16x32 f16) lane layout: a[j] = A[lq][half*8 + j], a[8+j] = A[lq][16 + half*8 + j]
        v16h a0;
        #pragma unroll
        for (int j = 0; j < 8; ++j) {
            a0[j]     = g0A[lq][half * 8 + j];
            a0[8 + j] = g0A[lq][16 + half * 8 + j];
        }
        v8f acc0[4];
        #pragma unroll
        for (int tile = 0; tile < 4; ++tile) {
            acc0[tile] = __builtin_amdgcn_wmma_f32_16x16x32_f16(
                false, a0, false, Wb0[tile], (short)0, bias0v[tile], false, false);
        }

        // elementwise LSTM cell (gate order i, f, g, o) + stage h0[t]
        #pragma unroll
        for (int v = 0; v < 8; ++v) {
            float ig = fsigmoid_half(acc0[0][v]);   // acc already = gate/2
            float fg = fsigmoid_half(acc0[1][v]);
            float gg = ftanh_fast   (acc0[2][v]);
            float og = fsigmoid_half(acc0[3][v]);
            float c  = fg * c0s[v] + ig * gg;
            c0s[v] = c;
            float h = og * ftanh_fast(c);
            const int M = v + half * 8;
            const _Float16 hh = (_Float16)h;
            g0A[M][4 + lq] = hh;   // h0[t] for next step's layer-0 A
            g1A[M][lq]     = hh;   // h0[t] for this step's layer-1 A
        }

        // ---- layer 1: gates1 = [h0[t] | h1[t-1]] @ [Wih1; Whh1]^T + bias ----
        v16h a1;  // must read h1[t-1] (cols 16..31) BEFORE overwriting below
        #pragma unroll
        for (int j = 0; j < 8; ++j) {
            a1[j]     = g1A[lq][half * 8 + j];
            a1[8 + j] = g1A[lq][16 + half * 8 + j];
        }
        v8f acc1[4];
        #pragma unroll
        for (int tile = 0; tile < 4; ++tile) {
            acc1[tile] = __builtin_amdgcn_wmma_f32_16x16x32_f16(
                false, a1, false, Wb1[tile], (short)0, bias1v[tile], false, false);
        }

        #pragma unroll
        for (int v = 0; v < 8; ++v) {
            float ig = fsigmoid_half(acc1[0][v]);
            float fg = fsigmoid_half(acc1[1][v]);
            float gg = ftanh_fast   (acc1[2][v]);
            float og = fsigmoid_half(acc1[3][v]);
            float c  = fg * c1s[v] + ig * gg;
            c1s[v] = c;
            float h = og * ftanh_fast(c);
            const int M = v + half * 8;
            g1A[M][16 + lq] = (_Float16)h;   // h1[t] (next step + FC below)
        }

        // ---- FC: y[b,t] = h1[t] . Wfc + bfc ----
        if (lane < 16) {
            float y = bfcv;
            #pragma unroll
            for (int k = 0; k < 16; ++k)
                y += (float)g1A[lq][16 + k] * wfcv[k];
            out[(size_t)(b0 + lq) * Tn + t] = y;
        }

        xv = xv_next;
    }
}

extern "C" void kernel_launch(void* const* d_in, const int* in_sizes, int n_in,
                              void* d_out, int out_size, void* d_ws, size_t ws_size,
                              hipStream_t stream) {
    const float* x    = (const float*)d_in[0];
    const float* Wih0 = (const float*)d_in[1];
    const float* Whh0 = (const float*)d_in[2];
    const float* bih0 = (const float*)d_in[3];
    const float* bhh0 = (const float*)d_in[4];
    const float* Wih1 = (const float*)d_in[5];
    const float* Whh1 = (const float*)d_in[6];
    const float* bih1 = (const float*)d_in[7];
    const float* bhh1 = (const float*)d_in[8];
    const float* Wfc  = (const float*)d_in[9];
    const float* bfc  = (const float*)d_in[10];
    float* out = (float*)d_out;

    dim3 grid(Bn / 16);   // 128 batch tiles, one wave each
    dim3 block(32);       // exactly one wave32
    hipLaunchKernelGGL(lstm2_fused_kernel, grid, block, 0, stream,
                       x, Wih0, Whh0, bih0, bhh0,
                       Wih1, Whh1, bih1, bhh1, Wfc, bfc, out);
}
